// SelfAttention_79310866088607
// MI455X (gfx1250) — compile-verified, matrix operand order checked by reading, not run on hardware
//
#include <hip/hip_runtime.h>
#include <hip/hip_bf16.h>

// ---------------------------------------------------------------------------
// Types / WMMA helpers (CDNA5 / gfx1250, wave32)
// ---------------------------------------------------------------------------
typedef __attribute__((ext_vector_type(16))) _Float16 v16h;
typedef __attribute__((ext_vector_type(8)))  float    v8f;

#define BATCH 4
#define CHN   512
#define NPIX  4096
#define CN    (CHN * NPIX)          // 2097152 elems per batch
#define EPSV  1e-5f

__device__ __forceinline__ v8f wmma_f16(v16h a, v16h b, v8f c) {
    return __builtin_amdgcn_wmma_f32_16x16x32_f16(false, a, false, b,
                                                  (short)0, c, false, false);
}

// A (16x32, MxK) / B^T (NxK) fragment pointer for row-major f16 source.
// Lane L: row = base_row + (L&15); k-group = 8*(L>>4).
// VGPR 0..3 <- k = kg+0..7 ; VGPR 4..7 <- k = kg+16..23  (two 16B loads).
__device__ __forceinline__ const _Float16*
frag_ptr(const _Float16* __restrict__ base, int row, int stride, int col) {
    int lane = threadIdx.x & 31;
    return base + (size_t)(row + (lane & 15)) * stride + col + ((lane >> 4) << 3);
}

__device__ __forceinline__ v16h frag_ld(const _Float16* __restrict__ p) {
    v16h a;
    *(uint4*)&a       = *(const uint4*)(p);
    *((uint4*)&a + 1) = *(const uint4*)(p + 16);
    return a;
}

// gfx1250 async global->LDS copy (16B per lane), tracked by ASYNCcnt.
__device__ __forceinline__ void async_b128_to_lds(const _Float16* gsrc, void* lds_dst) {
    unsigned           dst = (unsigned)(uintptr_t)lds_dst;          // low 32 bits = LDS offset
    unsigned long long src = (unsigned long long)(uintptr_t)gsrc;
    asm volatile("global_load_async_to_lds_b128 %0, %1, off"
                 :: "v"(dst), "v"(src) : "memory");
}

// ---------------------------------------------------------------------------
// 1) prep: zero stats, convert weights to f16
// ---------------------------------------------------------------------------
__global__ void sa_prep(const float* __restrict__ qkv_w,
                        const float* __restrict__ proj_w,
                        _Float16* __restrict__ w16q,
                        _Float16* __restrict__ w16p,
                        float* __restrict__ stats) {
    size_t i = (size_t)blockIdx.x * 256 + threadIdx.x;
    if (i < 8) stats[i] = 0.0f;
    if (i < (size_t)3 * CHN * CHN) w16q[i] = (_Float16)qkv_w[i];
    if (i < (size_t)CHN * CHN)     w16p[i] = (_Float16)proj_w[i];
}

// ---------------------------------------------------------------------------
// 2) GroupNorm stats: per-batch sum / sumsq
// ---------------------------------------------------------------------------
__global__ void sa_gn_stats(const float* __restrict__ x, float* __restrict__ stats) {
    const int b = blockIdx.y;
    const float* xb = x + (size_t)b * CN;
    float s = 0.0f, ss = 0.0f;
    for (int i = threadIdx.x + blockIdx.x * 256; i < CN; i += 512 * 256) {
        float v = xb[i];
        s += v; ss += v * v;
    }
    for (int m = 16; m; m >>= 1) { s += __shfl_xor(s, m); ss += __shfl_xor(ss, m); }
    __shared__ float sh[2][8];
    int w = threadIdx.x >> 5;
    if ((threadIdx.x & 31) == 0) { sh[0][w] = s; sh[1][w] = ss; }
    __syncthreads();
    if (threadIdx.x == 0) {
        float a = 0.0f, c2 = 0.0f;
        for (int i = 0; i < 8; ++i) { a += sh[0][i]; c2 += sh[1][i]; }
        atomicAdd(&stats[b * 2 + 0], a);
        atomicAdd(&stats[b * 2 + 1], c2);
    }
}

// ---------------------------------------------------------------------------
// 3) normalize + affine + transpose: x(b,c,n) f32 -> normedT(b,n,c) f16
// ---------------------------------------------------------------------------
__global__ void sa_gn_norm_t(const float* __restrict__ x,
                             const float* __restrict__ gn_w,
                             const float* __restrict__ gn_b,
                             const float* __restrict__ stats,
                             _Float16* __restrict__ nrmT) {
    __shared__ float tile[32][33];
    const int b  = blockIdx.z;
    const int p0 = blockIdx.x * 32;
    const int c0 = blockIdx.y * 32;
    const float mean = stats[b * 2 + 0] * (1.0f / (float)CN);
    const float var  = stats[b * 2 + 1] * (1.0f / (float)CN) - mean * mean;
    const float rstd = rsqrtf(var + EPSV);
    const int tx = threadIdx.x, ty = threadIdx.y;
    const float* xb = x + (size_t)b * CN;
    tile[ty][tx] = (xb[(size_t)(c0 + ty) * NPIX + p0 + tx] - mean) * rstd;
    __syncthreads();
    const int ch = c0 + tx;
    const float g = gn_w[ch], be = gn_b[ch];
    nrmT[((size_t)b * NPIX + p0 + ty) * CHN + ch] = (_Float16)(tile[tx][ty] * g + be);
}

// ---------------------------------------------------------------------------
// 4) QKV GEMM: qkv(1536,4096) = W(1536,512) x normed(512,4096) + bias
//    Block = 4 waves sharing one 64-row weight tile staged in LDS via
//    double-buffered async global->LDS copies (ASYNCcnt).  Wave w owns the
//    32-pixel strip [blockIdx.x*128 + 32w, +32).  q,k -> (b,n,c); v -> (b,c,n)
// ---------------------------------------------------------------------------
__global__ void __launch_bounds__(128)
sa_qkv_gemm(const _Float16* __restrict__ w16,
            const _Float16* __restrict__ nrmT,
            const float* __restrict__ qkv_b,
            _Float16* __restrict__ qbuf,
            _Float16* __restrict__ kbuf,
            _Float16* __restrict__ vbuf) {
    const int wave  = threadIdx.x >> 5;
    const int lane  = threadIdx.x & 31;
    const int hi    = lane >> 4;
    const int pBase = blockIdx.x * 128 + wave * 32;
    const int dBase = blockIdx.y * 64;
    const int b     = blockIdx.z;
    const _Float16* Bn = nrmT + (size_t)b * NPIX * CHN;

    __shared__ _Float16 ldsA[2][64][32];    // double-buffered 64x32 A tile (8KB)

    // stage one 64x32 f16 tile: 256 x 16B chunks, 2 per thread
    auto stage = [&](int buf, int kk) {
#pragma unroll
        for (int i = 0; i < 2; ++i) {
            const int cid = threadIdx.x * 2 + i;
            const int r   = cid >> 2;
            const int cc  = (cid & 3) * 8;
            async_b128_to_lds(w16 + (size_t)(dBase + r) * CHN + kk + cc,
                              &ldsA[buf][r][cc]);
        }
    };

    v8f acc[4][2] = {};
    stage(0, 0);
    for (int kk = 0; kk < CHN; kk += 32) {
        const int cur = (kk >> 5) & 1;
        if (kk + 32 < CHN) {
            stage(cur ^ 1, kk + 32);
            asm volatile("s_wait_asynccnt 0x2" ::: "memory");   // our 2 older copies done
        } else {
            asm volatile("s_wait_asynccnt 0x0" ::: "memory");
        }
        __syncthreads();                                        // everyone's tile visible

        v16h bf0 = frag_ld(frag_ptr(Bn, pBase,      CHN, kk));
        v16h bf1 = frag_ld(frag_ptr(Bn, pBase + 16, CHN, kk));
#pragma unroll
        for (int mt = 0; mt < 4; ++mt) {
            const _Float16* ap = &ldsA[cur][mt * 16 + (lane & 15)][hi * 8];
            v16h af;
            *(uint4*)&af       = *(const uint4*)(ap);
            *((uint4*)&af + 1) = *(const uint4*)(ap + 16);
            acc[mt][0] = wmma_f16(af, bf0, acc[mt][0]);
            acc[mt][1] = wmma_f16(af, bf1, acc[mt][1]);
        }
        // no tail barrier: next overwrite of ldsA[cur] (kk+64 data) is issued at the top of
        // iteration kk+32, before its barrier, so it is not ordered after other waves'
        // reads of ldsA[cur] in this iteration's compute.
    }

    const int sel = dBase >> 9;            // 0: q, 1: k, 2: v
#pragma unroll
    for (int mt = 0; mt < 4; ++mt) {
        const int d0 = dBase + mt * 16 + hi * 8;
        float bias[8];
#pragma unroll
        for (int j = 0; j < 8; ++j) bias[j] = qkv_b[d0 + j];
#pragma unroll
        for (int nt = 0; nt < 2; ++nt) {
            const int p = pBase + nt * 16 + (lane & 15);
            if (sel < 2) {
                alignas(16) _Float16 h[8];
#pragma unroll
                for (int j = 0; j < 8; ++j)
                    h[j] = (_Float16)(acc[mt][nt][j] + bias[j]);
                _Float16* dst = (sel == 0 ? qbuf : kbuf) +
                    ((size_t)b * NPIX + p) * CHN + (d0 - sel * CHN);
                *(uint4*)dst = *(uint4*)h;
            } else {
                const int dloc = d0 - 2 * CHN;
#pragma unroll
                for (int j = 0; j < 8; ++j)
                    vbuf[((size_t)b * CHN + dloc + j) * NPIX + p] =
                        (_Float16)(acc[mt][nt][j] + bias[j]);
            }
        }
    }
}

// ---------------------------------------------------------------------------
// 5) Flash attention + residual, KV block = 64 rows.
//    Block = 4 waves, one 16-row query tile; wave w owns channels [128w,128w+128).
//    Partial S reduced across waves in LDS; online softmax replicated; P goes
//    through LDS to become the A operand for PV.  Writes resT = O + x  (b,n,c) f16.
// ---------------------------------------------------------------------------
__global__ void __launch_bounds__(128)
sa_attn_flash(const _Float16* __restrict__ qbuf,
              const _Float16* __restrict__ kbuf,
              const _Float16* __restrict__ vbuf,
              const float* __restrict__ x,
              _Float16* __restrict__ resT) {
    const int pBase = blockIdx.x * 16;
    const int b     = blockIdx.y;
    const int wave  = threadIdx.x >> 5;
    const int lane  = threadIdx.x & 31;
    const int hi    = lane >> 4;
    const int cOff  = wave * 128;

    const _Float16* Q = qbuf + (size_t)b * NPIX * CHN;
    const _Float16* K = kbuf + (size_t)b * NPIX * CHN;
    const _Float16* V = vbuf + (size_t)b * CHN * NPIX;

    __shared__ float    ldsS[4][4][8][32];   // [wave][kv-subtile][slot][lane] 16KB
    __shared__ _Float16 ldsP[16][64];        // P tile (16 x 64)                2KB

    v16h qf[4];
#pragma unroll
    for (int kc = 0; kc < 4; ++kc)
        qf[kc] = frag_ld(frag_ptr(Q, pBase, CHN, cOff + kc * 32));

    v8f acc[8] = {};
    float mrun[8], lrun[8];
#pragma unroll
    for (int j = 0; j < 8; ++j) { mrun[j] = -3.0e38f; lrun[j] = 0.0f; }

    const float scale = 0.04419417382415922f;   // 1/sqrt(512)

    for (int jb = 0; jb < NPIX; jb += 64) {
        // ---- partial S (16 x 64) over this wave's 128 channels ----
        v8f s[4] = {};
#pragma unroll
        for (int kc = 0; kc < 4; ++kc) {
#pragma unroll
            for (int t = 0; t < 4; ++t) {
                v16h kf = frag_ld(frag_ptr(K, jb + t * 16, CHN, cOff + kc * 32));
                s[t] = wmma_f16(qf[kc], kf, s[t]);
            }
        }
#pragma unroll
        for (int t = 0; t < 4; ++t)
#pragma unroll
            for (int j = 0; j < 8; ++j) ldsS[wave][t][j][lane] = s[t][j];
        __syncthreads();

        // ---- full S (replicated across waves), online softmax ----
        float f[4][8], al[8];
#pragma unroll
        for (int t = 0; t < 4; ++t)
#pragma unroll
            for (int j = 0; j < 8; ++j)
                f[t][j] = (ldsS[0][t][j][lane] + ldsS[1][t][j][lane] +
                           ldsS[2][t][j][lane] + ldsS[3][t][j][lane]) * scale;
#pragma unroll
        for (int j = 0; j < 8; ++j) {
            float mx = fmaxf(fmaxf(f[0][j], f[1][j]), fmaxf(f[2][j], f[3][j]));
            mx = fmaxf(mx, __shfl_xor(mx, 1));
            mx = fmaxf(mx, __shfl_xor(mx, 2));
            mx = fmaxf(mx, __shfl_xor(mx, 4));
            mx = fmaxf(mx, __shfl_xor(mx, 8));
            float mnew = fmaxf(mrun[j], mx);
            al[j] = __expf(mrun[j] - mnew);
            float r = 0.0f;
#pragma unroll
            for (int t = 0; t < 4; ++t) {
                f[t][j] = __expf(f[t][j] - mnew);
                r += f[t][j];
            }
            r += __shfl_xor(r, 1);
            r += __shfl_xor(r, 2);
            r += __shfl_xor(r, 4);
            r += __shfl_xor(r, 8);
            lrun[j] = lrun[j] * al[j] + r;
            mrun[j] = mnew;
        }
        // rescale running accumulator
#pragma unroll
        for (int t = 0; t < 8; ++t)
#pragma unroll
            for (int j = 0; j < 8; ++j) acc[t][j] *= al[j];

        // ---- stage P (16x64) through LDS to re-fragment as A operand ----
        if (wave == 0) {
#pragma unroll
            for (int t = 0; t < 4; ++t)
#pragma unroll
                for (int j = 0; j < 8; ++j)
                    ldsP[hi * 8 + j][t * 16 + (lane & 15)] = (_Float16)f[t][j];
        }
        __syncthreads();

        v16h pf0, pf1;
        {
            const _Float16* pp = &ldsP[lane & 15][hi * 8];
            *(uint4*)&pf0       = *(const uint4*)(pp);
            *((uint4*)&pf0 + 1) = *(const uint4*)(pp + 16);
            *(uint4*)&pf1       = *(const uint4*)(pp + 32);
            *((uint4*)&pf1 + 1) = *(const uint4*)(pp + 48);
        }

        // ---- O += P x V over this wave's channel slice ----
#pragma unroll
        for (int nt = 0; nt < 8; ++nt) {
            v16h vf0 = frag_ld(frag_ptr(V, cOff + nt * 16, NPIX, jb));
            v16h vf1 = frag_ld(frag_ptr(V, cOff + nt * 16, NPIX, jb + 32));
            acc[nt] = wmma_f16(pf0, vf0, acc[nt]);
            acc[nt] = wmma_f16(pf1, vf1, acc[nt]);
        }
        __syncthreads();
    }

    // ---- epilogue: O / l + x  ->  resT (b,n,c) f16 ----
    float inv[8];
#pragma unroll
    for (int j = 0; j < 8; ++j) inv[j] = 1.0f / lrun[j];
    const float* xb = x + (size_t)b * CN;
    _Float16* R = resT + (size_t)b * NPIX * CHN;
#pragma unroll
    for (int nt = 0; nt < 8; ++nt) {
        const int ch = cOff + nt * 16 + (lane & 15);
        const float* xp = xb + (size_t)ch * NPIX + pBase + hi * 8;
#pragma unroll
        for (int j = 0; j < 8; ++j) {
            float o = acc[nt][j] * inv[j] + xp[j];
            R[(size_t)(pBase + hi * 8 + j) * CHN + ch] = (_Float16)o;
        }
    }
}

// ---------------------------------------------------------------------------
// 6) Proj GEMM: out(b,512,4096) f32 = proj_w(512,512) x res(512,4096) + bias
//    Same 4-wave / async-LDS-staged-A structure as the QKV GEMM.
// ---------------------------------------------------------------------------
__global__ void __launch_bounds__(128)
sa_proj_gemm(const _Float16* __restrict__ w16,
             const _Float16* __restrict__ resT,
             const float* __restrict__ proj_b,
             float* __restrict__ out) {
    const int wave  = threadIdx.x >> 5;
    const int lane  = threadIdx.x & 31;
    const int hi    = lane >> 4;
    const int pBase = blockIdx.x * 128 + wave * 32;
    const int dBase = blockIdx.y * 64;
    const int b     = blockIdx.z;
    const _Float16* Bn = resT + (size_t)b * NPIX * CHN;

    __shared__ _Float16 ldsA[2][64][32];

    auto stage = [&](int buf, int kk) {
#pragma unroll
        for (int i = 0; i < 2; ++i) {
            const int cid = threadIdx.x * 2 + i;
            const int r   = cid >> 2;
            const int cc  = (cid & 3) * 8;
            async_b128_to_lds(w16 + (size_t)(dBase + r) * CHN + kk + cc,
                              &ldsA[buf][r][cc]);
        }
    };

    v8f acc[4][2] = {};
    stage(0, 0);
    for (int kk = 0; kk < CHN; kk += 32) {
        const int cur = (kk >> 5) & 1;
        if (kk + 32 < CHN) {
            stage(cur ^ 1, kk + 32);
            asm volatile("s_wait_asynccnt 0x2" ::: "memory");
        } else {
            asm volatile("s_wait_asynccnt 0x0" ::: "memory");
        }
        __syncthreads();

        v16h bf0 = frag_ld(frag_ptr(Bn, pBase,      CHN, kk));
        v16h bf1 = frag_ld(frag_ptr(Bn, pBase + 16, CHN, kk));
#pragma unroll
        for (int mt = 0; mt < 4; ++mt) {
            const _Float16* ap = &ldsA[cur][mt * 16 + (lane & 15)][hi * 8];
            v16h af;
            *(uint4*)&af       = *(const uint4*)(ap);
            *((uint4*)&af + 1) = *(const uint4*)(ap + 16);
            acc[mt][0] = wmma_f16(af, bf0, acc[mt][0]);
            acc[mt][1] = wmma_f16(af, bf1, acc[mt][1]);
        }
    }
#pragma unroll
    for (int mt = 0; mt < 4; ++mt) {
        const int d0 = dBase + mt * 16 + hi * 8;
#pragma unroll
        for (int nt = 0; nt < 2; ++nt) {
            const int p = pBase + nt * 16 + (lane & 15);
#pragma unroll
            for (int j = 0; j < 8; ++j)
                out[((size_t)b * CHN + d0 + j) * NPIX + p] =
                    acc[mt][nt][j] + proj_b[d0 + j];
        }
    }
}

// ---------------------------------------------------------------------------
// Launch
// ---------------------------------------------------------------------------
extern "C" void kernel_launch(void* const* d_in, const int* in_sizes, int n_in,
                              void* d_out, int out_size, void* d_ws, size_t ws_size,
                              hipStream_t stream) {
    (void)in_sizes; (void)n_in; (void)out_size; (void)ws_size;
    const float* x      = (const float*)d_in[0];
    const float* gn_w   = (const float*)d_in[1];
    const float* gn_b   = (const float*)d_in[2];
    const float* qkv_w  = (const float*)d_in[3];
    const float* qkv_b  = (const float*)d_in[4];
    const float* proj_w = (const float*)d_in[5];
    const float* proj_b = (const float*)d_in[6];
    float* out = (float*)d_out;

    char* ws = (char*)d_ws;
    size_t off = 0;
    auto alloc = [&](size_t bytes) -> void* {
        off = (off + 255) & ~(size_t)255;
        void* p = ws + off;
        off += bytes;
        return p;
    };
    float*    stats = (float*)   alloc(8 * sizeof(float));
    _Float16* w16q  = (_Float16*)alloc((size_t)3 * CHN * CHN * 2);
    _Float16* w16p  = (_Float16*)alloc((size_t)CHN * CHN * 2);
    _Float16* nrmT  = (_Float16*)alloc((size_t)BATCH * NPIX * CHN * 2);
    _Float16* qbuf  = (_Float16*)alloc((size_t)BATCH * NPIX * CHN * 2);
    _Float16* kbuf  = (_Float16*)alloc((size_t)BATCH * NPIX * CHN * 2);
    _Float16* vbuf  = (_Float16*)alloc((size_t)BATCH * CHN * NPIX * 2);
    _Float16* rbuf  = (_Float16*)alloc((size_t)BATCH * NPIX * CHN * 2);

    // 1) prep (zero stats + weight conversion)
    sa_prep<<<dim3((3 * CHN * CHN + 255) / 256), 256, 0, stream>>>(
        qkv_w, proj_w, w16q, w16p, stats);
    // 2) GroupNorm stats
    sa_gn_stats<<<dim3(512, BATCH), 256, 0, stream>>>(x, stats);
    // 3) normalize + transpose to (b,n,c) f16
    sa_gn_norm_t<<<dim3(NPIX / 32, CHN / 32, BATCH), dim3(32, 32), 0, stream>>>(
        x, gn_w, gn_b, stats, nrmT);
    // 4) QKV projection (4 waves, async-staged A tile)
    sa_qkv_gemm<<<dim3(NPIX / 128, (3 * CHN) / 64, BATCH), 128, 0, stream>>>(
        w16q, nrmT, qkv_b, qbuf, kbuf, vbuf);
    // 5) flash attention + residual (KV block = 64)
    sa_attn_flash<<<dim3(NPIX / 16, BATCH), 128, 0, stream>>>(
        qbuf, kbuf, vbuf, x, rbuf);
    // 6) output projection (4 waves, async-staged A tile)
    sa_proj_gemm<<<dim3(NPIX / 128, CHN / 64, BATCH), 128, 0, stream>>>(
        w16p, rbuf, proj_b, out);
}
